// MultiHeadAttention_52269751992827
// MI455X (gfx1250) — matrix operand order repaired and checked by reading, not compiled, not run on hardware
//
#include <hip/hip_runtime.h>
#include <hip/hip_bf16.h>

typedef __attribute__((ext_vector_type(16))) _Float16 v16h;
typedef __attribute__((ext_vector_type(8)))  float    v8f;
typedef __attribute__((ext_vector_type(4)))  unsigned int u32x4;
typedef __attribute__((ext_vector_type(8)))  int      i32x8;
typedef __attribute__((ext_vector_type(4)))  int      i32x4;

#define B_    8
#define S_    2048
#define H_    8
#define DK_   64
#define DM_   512
#define HD_   512            // H_*DK_
#define NROW_ 16384          // B_*S_
#define NEGV  (-2147483648.0f)
#define CHUNK 64             // keys per pipeline stage

static __device__ __forceinline__ int lane_id() { return threadIdx.x & 31; }

// ---- WMMA fragment loaders (wave32, contiguous k order k = 16*(lane/16)+i) ----------

// A matrix 16x32 f16, row-major source, row stride ld.
// lane<16: row=lane,   K = 0..15
// lane>=16:row=lane-16,K = 16..31
static __device__ __forceinline__ v16h a_frag_h(const _Float16* base, int ld) {
  const int l = lane_id();
  const _Float16* p = base + (size_t)(l & 15) * ld + ((l >> 4) << 4);
  v16h a;
#pragma unroll
  for (int i = 0; i < 8; ++i) { a[i] = p[i]; a[8 + i] = p[8 + i]; }
  return a;
}

static __device__ __forceinline__ v16h a_frag_f(const float* base, int ld) {
  const int l = lane_id();
  const float* p = base + (size_t)(l & 15) * ld + ((l >> 4) << 4);
  v16h a;
#pragma unroll
  for (int i = 0; i < 8; ++i) { a[i] = (_Float16)p[i]; a[8 + i] = (_Float16)p[8 + i]; }
  return a;
}

// B matrix 32x16 f16, element (k,n) = base[k*ld + n].
// lane<16: col=lane, K=0..15 ; lane>=16: col=lane-16, K=16..31
static __device__ __forceinline__ v16h b_frag_h(const _Float16* base, int ld) {
  const int l = lane_id();
  const int n = l & 15, kb = (l >> 4) << 4;
  v16h b;
#pragma unroll
  for (int i = 0; i < 16; ++i) b[i] = base[(size_t)(kb + i) * ld + n];
  return b;
}

static __device__ __forceinline__ v16h b_frag_f(const float* base, int ld) {
  const int l = lane_id();
  const int n = l & 15, kb = (l >> 4) << 4;
  v16h b;
#pragma unroll
  for (int i = 0; i < 16; ++i) b[i] = (_Float16)base[(size_t)(kb + i) * ld + n];
  return b;
}

// B = K^T : element (k,n) = Ksrc[n*ld + k]  -> contiguous 16-half loads
static __device__ __forceinline__ v16h bT_frag_h(const _Float16* base, int ld) {
  const int l = lane_id();
  const _Float16* p = base + (size_t)(l & 15) * ld + ((l >> 4) << 4);
  v16h b;
#pragma unroll
  for (int i = 0; i < 16; ++i) b[i] = p[i];
  return b;
}

static __device__ __forceinline__ v8f wmma_f(v16h a, v16h b, v8f c) {
  return __builtin_amdgcn_wmma_f32_16x16x32_f16(false, a, false, b, (short)0, c,
                                                false, false);
}

// ---- Tensor Data Mover: 2-D f16 tile (tile_w x tile_h) global -> LDS ------
// D# packing per CDNA5 ISA 8.3/8.4: group0 = {flags, lds_addr, global_addr,
// type=2}; group1 = {data_size=2B, tensor dims, tile dims, row stride}.
static __device__ __forceinline__ void tdm_load_2d_f16(
    unsigned lds_addr, const void* gptr,
    unsigned tile_w, unsigned tile_h, unsigned row_stride_elems) {
  const unsigned long long ga = (unsigned long long)gptr;
  u32x4 g0;
  g0[0] = 1u;                                        // count=1, user descriptor
  g0[1] = lds_addr;                                  // LDS byte address
  g0[2] = (unsigned)(ga & 0xFFFFFFFFu);              // global_addr[31:0]
  g0[3] = (unsigned)((ga >> 32) & 0x01FFFFFFu) | (2u << 30);  // addr[56:32]|type=2
  i32x8 g1;
  g1[0] = (int)(1u << 16);                           // data_size=1 -> 2 bytes
  g1[1] = (int)((HD_ & 0xFFFFu) << 16);              // tensor_dim0[15:0]
  g1[2] = (int)(((unsigned)HD_ >> 16) | ((NROW_ & 0xFFFFu) << 16)); // dim0 hi|dim1 lo
  g1[3] = (int)(((unsigned)NROW_ >> 16) | (tile_w << 16));          // dim1 hi|tile_dim0
  g1[4] = (int)(tile_h & 0xFFFFu);                   // tile_dim1 (tile_dim2=0)
  g1[5] = (int)row_stride_elems;                     // tensor_dim0_stride[31:0]
  g1[6] = 0;                                         // stride[47:32]
  g1[7] = 0;
  const i32x4 z4 = {0, 0, 0, 0};
#if __clang_major__ >= 23
  const i32x8 z8 = {0, 0, 0, 0, 0, 0, 0, 0};
  __builtin_amdgcn_tensor_load_to_lds(g0, g1, z4, z4, z8, 0);
#else
  __builtin_amdgcn_tensor_load_to_lds(g0, g1, z4, z4, 0);
#endif
}

static __device__ __forceinline__ unsigned lds_addr_of(const void* p) {
  // LDS aperture: low 32 bits of the generic address are the LDS byte address
  return (unsigned)(unsigned long long)p;
}

// ---- Kernel 1: shared projection (x @ Wq + bq) for q,k,v -> f16 planes ----
__global__ void proj_kernel(const float* __restrict__ q, const float* __restrict__ k,
                            const float* __restrict__ v, const float* __restrict__ Wq,
                            const float* __restrict__ bq, _Float16* __restrict__ ws) {
  const int wave = threadIdx.x >> 5;
  const int tile = blockIdx.x * 4 + wave;        // 32768 tiles total
  const int m0 = (tile >> 5) * 16;               // 1024 M tiles
  const int n0 = (tile & 31) * 16;               // 32 N tiles
  const float* x = (blockIdx.y == 0) ? q : (blockIdx.y == 1) ? k : v;
  _Float16* dst = ws + (size_t)blockIdx.y * (size_t)NROW_ * HD_;

  v8f c = {};
#pragma unroll
  for (int kk = 0; kk < DK_; kk += 32) {
    v16h a = a_frag_f(x + (size_t)m0 * DK_ + kk, DK_);
    v16h b = b_frag_f(Wq + (size_t)kk * HD_ + n0, HD_);
    c = wmma_f(a, b, c);
  }
  const int l = lane_id(), n = l & 15, hi = l >> 4;
  const float bias = bq[n0 + n];
#pragma unroll
  for (int r = 0; r < 8; ++r) {
    const int m = r + 8 * hi;
    dst[(size_t)(m0 + m) * HD_ + n0 + n] = (_Float16)(c[r] + bias);
  }
}

// ---- Kernel 2: TDM-fed, double-buffered flash attention -------------------
// Block = 4 waves; each wave owns one 16-row q tile; all waves share the
// K/V chunk (64 keys x 64 dk, f16) staged in LDS by the Tensor Data Mover.
__global__ void flash_attn_kernel(const _Float16* __restrict__ Qp,
                                  const _Float16* __restrict__ Kp,
                                  const _Float16* __restrict__ Vp,
                                  const float* __restrict__ mask,
                                  _Float16* __restrict__ Yw) {
  __shared__ _Float16 sK[2][CHUNK][DK_];   // 16 KB
  __shared__ _Float16 sV[2][CHUNK][DK_];   // 16 KB
  __shared__ _Float16 sP[4][16][CHUNK];    // 8 KB, per-wave P staging

  const int wave = threadIdx.x >> 5;
  const int l = lane_id(), n = l & 15, hi = l >> 4;
  const int q0 = (blockIdx.x * 4 + wave) * 16;
  const int h  = blockIdx.y & (H_ - 1);
  const int b  = blockIdx.y >> 3;

  const size_t rowbase = (size_t)b * S_;
  const _Float16* Qb = Qp + (rowbase + q0) * HD_ + h * DK_;
  const float*    mb = mask + ((size_t)b * S_ + q0) * S_;

  // Q fragments persist in registers (DK=64 -> two K chunks)
  const v16h aq0 = a_frag_h(Qb + 0,  HD_);
  const v16h aq1 = a_frag_h(Qb + 32, HD_);

  v8f acc[4];
#pragma unroll
  for (int t = 0; t < 4; ++t) acc[t] = (v8f){};
  float rmax[8], rsum[8];
#pragma unroll
  for (int r = 0; r < 8; ++r) { rmax[r] = -__builtin_inff(); rsum[r] = 0.0f; }

  // Prime the pipeline: wave 0 DMAs chunk 0 into buffer 0.
  if (threadIdx.x < 32) {
    tdm_load_2d_f16(lds_addr_of(&sK[0][0][0]),
                    Kp + rowbase * HD_ + h * DK_, DK_, CHUNK, HD_);
    tdm_load_2d_f16(lds_addr_of(&sV[0][0][0]),
                    Vp + rowbase * HD_ + h * DK_, DK_, CHUNK, HD_);
    __builtin_amdgcn_s_wait_tensorcnt(0);
  }
  __syncthreads();

  for (int kc = 0; kc < S_; kc += CHUNK) {
    const int buf = (kc / CHUNK) & 1;
    const bool pf = (kc + CHUNK) < S_;

    // Prefetch next chunk into the other buffer while we compute.
    if (pf && threadIdx.x < 32) {
      tdm_load_2d_f16(lds_addr_of(&sK[buf ^ 1][0][0]),
                      Kp + (rowbase + kc + CHUNK) * HD_ + h * DK_, DK_, CHUNK, HD_);
      tdm_load_2d_f16(lds_addr_of(&sV[buf ^ 1][0][0]),
                      Vp + (rowbase + kc + CHUNK) * HD_ + h * DK_, DK_, CHUNK, HD_);
    }

    // ---- scores: four 16x16 tiles over 64 key columns (K^T from LDS) ----
    v8f cs[4];
#pragma unroll
    for (int t = 0; t < 4; ++t) {
      v8f c = {};
      c = wmma_f(aq0, bT_frag_h(&sK[buf][16 * t][0],  DK_), c);
      c = wmma_f(aq1, bT_frag_h(&sK[buf][16 * t][32], DK_), c);
      cs[t] = c;
    }

    // ---- scale + mask (fill NEG where mask==1) ----
    float sv[4][8];
#pragma unroll
    for (int t = 0; t < 4; ++t)
#pragma unroll
      for (int r = 0; r < 8; ++r) {
        const int m = r + 8 * hi;
        const float mv = mb[(size_t)m * S_ + kc + 16 * t + n];
        sv[t][r] = (mv == 1.0f) ? NEGV : cs[t][r] * 0.125f;
      }

    // ---- online softmax: reductions across 16-lane column groups ----
    float alpha[8];
#pragma unroll
    for (int r = 0; r < 8; ++r) {
      float cm = fmaxf(fmaxf(sv[0][r], sv[1][r]), fmaxf(sv[2][r], sv[3][r]));
#pragma unroll
      for (int off = 8; off >= 1; off >>= 1) cm = fmaxf(cm, __shfl_xor(cm, off, 32));
      const float nm = fmaxf(rmax[r], cm);
      alpha[r] = __expf(rmax[r] - nm);
      rmax[r] = nm;
      const int m = r + 8 * hi;
      float ps = 0.0f;
#pragma unroll
      for (int t = 0; t < 4; ++t) {
        const float p = __expf(sv[t][r] - nm);
        ps += p;
        sP[wave][m][16 * t + n] = (_Float16)p;
      }
#pragma unroll
      for (int off = 8; off >= 1; off >>= 1) ps += __shfl_xor(ps, off, 32);
      rsum[r] = rsum[r] * alpha[r] + ps;
    }

    // P tile LDS write -> A-fragment read (same wave; DS pipe in-order,
    // order the VGPR results explicitly)
    __asm__ volatile("s_wait_dscnt 0x0" ::: "memory");
    const v16h ap0 = a_frag_h(&sP[wave][0][0],  CHUNK);
    const v16h ap1 = a_frag_h(&sP[wave][0][32], CHUNK);

    // ---- y = y*alpha + P @ V (V from LDS) ----
#pragma unroll
    for (int t = 0; t < 4; ++t) {
      v8f a = acc[t];
#pragma unroll
      for (int r = 0; r < 8; ++r) a[r] *= alpha[r];
      a = wmma_f(ap0, b_frag_h(&sV[buf][0][16 * t],  DK_), a);
      a = wmma_f(ap1, b_frag_h(&sV[buf][32][16 * t], DK_), a);
      acc[t] = a;
    }

    // Ensure the prefetch landed and all waves are done with this buffer.
    if (pf && threadIdx.x < 32) __builtin_amdgcn_s_wait_tensorcnt(0);
    __syncthreads();
  }

  _Float16* Yb = Yw + (rowbase + q0) * HD_ + h * DK_;
#pragma unroll
  for (int r = 0; r < 8; ++r) {
    const int m = r + 8 * hi;
    const float inv = 1.0f / rsum[r];
#pragma unroll
    for (int t = 0; t < 4; ++t)
      Yb[(size_t)m * HD_ + 16 * t + n] = (_Float16)(acc[t][r] * inv);
  }
}

// ---- Kernel 3: output projection Y @ Wo + bo -> f32 ----------------------
__global__ void outproj_kernel(const _Float16* __restrict__ Yw,
                               const float* __restrict__ Wo,
                               const float* __restrict__ bo,
                               float* __restrict__ out) {
  const int wave = threadIdx.x >> 5;
  const int tile = blockIdx.x * 4 + wave;
  const int m0 = (tile >> 5) * 16;
  const int n0 = (tile & 31) * 16;

  v8f c = {};
#pragma unroll 4
  for (int kk = 0; kk < HD_; kk += 32) {
    v16h a = a_frag_h(Yw + (size_t)m0 * HD_ + kk, HD_);
    v16h b = b_frag_f(Wo + (size_t)kk * DM_ + n0, DM_);
    c = wmma_f(a, b, c);
  }
  const int l = lane_id(), n = l & 15, hi = l >> 4;
  const float bias = bo[n0 + n];
#pragma unroll
  for (int r = 0; r < 8; ++r) {
    const int m = r + 8 * hi;
    out[(size_t)(m0 + m) * DM_ + n0 + n] = c[r] + bias;
  }
}

extern "C" void kernel_launch(void* const* d_in, const int* in_sizes, int n_in,
                              void* d_out, int out_size, void* d_ws, size_t ws_size,
                              hipStream_t stream) {
  (void)in_sizes; (void)n_in; (void)out_size; (void)ws_size;
  const float* q    = (const float*)d_in[0];
  const float* k    = (const float*)d_in[1];
  const float* v    = (const float*)d_in[2];
  const float* mask = (const float*)d_in[3];
  const float* Wq   = (const float*)d_in[4];
  const float* bq   = (const float*)d_in[5];
  const float* Wo   = (const float*)d_in[6];
  const float* bo   = (const float*)d_in[7];

  const size_t plane = (size_t)NROW_ * HD_;   // 8.4M f16 per plane
  _Float16* Qp = (_Float16*)d_ws;
  _Float16* Kp = Qp + plane;
  _Float16* Vp = Qp + 2 * plane;
  _Float16* Yw = Qp + 3 * plane;

  // 1) q/k/v projections: 32768 16x16 tiles each, 4 waves/block
  proj_kernel<<<dim3(8192, 3), 128, 0, stream>>>(q, k, v, Wq, bq, Qp);
  // 2) flash attention: 128 q-tiles / (h,b); 4 waves/block
  flash_attn_kernel<<<dim3(S_ / 64, H_ * B_), 128, 0, stream>>>(Qp, Kp, Vp, mask, Yw);
  // 3) output projection
  outproj_kernel<<<dim3(8192), 128, 0, stream>>>(Yw, Wo, bo, (float*)d_out);
}
